// NeuroCore_67319317397843
// MI455X (gfx1250) — matrix operand order repaired and checked by reading, not compiled, not run on hardware
//
#include <hip/hip_runtime.h>

// ---------------------------------------------------------------------------
// NeuroSAT-style message passing for MI455X (gfx1250, wave32, WMMA bf16)
// D = 128 fixed. GEMMs via v_wmma_f32_16x16x32_bf16 (f32 accumulate).
// Edge scatter via native f32 global atomics. All launches on `stream`.
// ---------------------------------------------------------------------------

typedef __bf16 bf16_t;
typedef __attribute__((ext_vector_type(16))) __bf16 v16bf;
typedef __attribute__((ext_vector_type(8)))  float  v8f;

union Frag { v16bf v; uint4 q[2]; };

#define D_DIM      128
#define N_VARS     65536
#define N_LITS     131072
#define N_CLAUSES  262144
#define N_ROUNDS   8

// ------------------------------- init --------------------------------------
__global__ __launch_bounds__(256) void fill_kernel(float* __restrict__ p,
                                                   const float* __restrict__ sp,
                                                   unsigned long long n) {
    unsigned long long i = (unsigned long long)blockIdx.x * 256ull + threadIdx.x;
    if (i < n) p[i] = sp[0];
}

// ------------------------- weight transpose to bf16 ------------------------
// W is (K x N) row-major fp32 -> Wt is (N x K) row-major bf16
__global__ __launch_bounds__(256) void wconv_kernel(const float* __restrict__ W,
                                                    bf16_t* __restrict__ Wt,
                                                    int K, int N) {
    int i = blockIdx.x * 256 + threadIdx.x;
    if (i >= K * N) return;
    int k = i / N, n = i % N;
    Wt[(size_t)n * K + k] = (bf16_t)W[(size_t)k * N + n];
}

// ------------------------------ edge scatter -------------------------------
// dst[didx[e]] += src[sidx[e]]   (one wave per edge, 4 floats per lane)
__global__ __launch_bounds__(256) void edge_scatter_kernel(
        const float* __restrict__ src, const int* __restrict__ sidx,
        const int* __restrict__ didx, float* __restrict__ dst, int n_edges) {
    int wave = (int)((blockIdx.x * 256u + threadIdx.x) >> 5);
    int lane = threadIdx.x & 31;
    if (wave >= n_edges) return;
    int s = sidx[wave];
    int d = didx[wave];
    const float4 v = *(const float4*)(src + (size_t)s * D_DIM + lane * 4);
    float* p = dst + (size_t)d * D_DIM + lane * 4;
    __hip_atomic_fetch_add(p + 0, v.x, __ATOMIC_RELAXED, __HIP_MEMORY_SCOPE_AGENT);
    __hip_atomic_fetch_add(p + 1, v.y, __ATOMIC_RELAXED, __HIP_MEMORY_SCOPE_AGENT);
    __hip_atomic_fetch_add(p + 2, v.z, __ATOMIC_RELAXED, __HIP_MEMORY_SCOPE_AGENT);
    __hip_atomic_fetch_add(p + 3, v.w, __ATOMIC_RELAXED, __HIP_MEMORY_SCOPE_AGENT);
}

// ------------------------------ fused MLP ----------------------------------
// Y <- relu([Y | msgs*scale (| flip Y)] @ W1 + b1) @ W2 + b2 + Y   (in place)
// Tile: 64 rows per block, 256 threads = 8 waves.
// Wave w: rows (w>>1)*16..+15, cols (w&1)*64..+63  (4 N-tiles of 16).
template<int KX, bool FLIP>
__global__ __launch_bounds__(256) void mlp_update_kernel(
        float* __restrict__ Y, const float* __restrict__ msgs,
        const float* __restrict__ scale_p,
        const bf16_t* __restrict__ W1t, const float* __restrict__ b1,
        const bf16_t* __restrict__ W2t, const float* __restrict__ b2) {
    constexpr int XS = KX + 8;     // padded stride (bank-conflict-free frags)
    constexpr int HS = 136;        // 128 + 8
    __shared__ __align__(16) bf16_t smem[64 * XS];

    const int tid     = threadIdx.x;
    const int rowBase = blockIdx.x * 64;
    const float scale = scale_p[0];

    // ---- stage X tile (bf16) ----
    for (int i = tid; i < 64 * KX; i += 256) {
        int r = i / KX, c = i % KX;
        int g = rowBase + r;
        float v;
        if (c < D_DIM)            v = Y[(size_t)g * D_DIM + c];
        else if (c < 2 * D_DIM)   v = msgs[(size_t)g * D_DIM + (c - D_DIM)] * scale;
        else                      v = Y[(size_t)(g ^ 1) * D_DIM + (c - 2 * D_DIM)];
        smem[r * XS + c] = (bf16_t)v;
    }
    __syncthreads();

    const int lane = tid & 31;
    const int wv   = tid >> 5;
    const int rq   = wv >> 1;       // row quarter
    const int ch   = wv & 1;        // column half
    const int lh   = lane >> 4;     // lane half
    const int ln   = lane & 15;
    const int am   = rq * 16 + ln;  // A-fragment row within tile

    // ---- layer 1 ----
    v8f acc[4];
#pragma unroll
    for (int nt = 0; nt < 4; ++nt) {
        float bv = b1[ch * 64 + nt * 16 + ln];
#pragma unroll
        for (int j = 0; j < 8; ++j) acc[nt][j] = bv;
    }
#pragma unroll 2
    for (int k0 = 0; k0 < KX; k0 += 32) {
        Frag a;
        a.q[0] = *(const uint4*)&smem[am * XS + k0 + lh * 8];
        a.q[1] = *(const uint4*)&smem[am * XS + k0 + 16 + lh * 8];
#pragma unroll
        for (int nt = 0; nt < 4; ++nt) {
            Frag bm;
            const bf16_t* wp = W1t + (size_t)(ch * 64 + nt * 16 + ln) * KX + k0 + lh * 8;
            bm.q[0] = *(const uint4*)(wp);
            bm.q[1] = *(const uint4*)(wp + 16);
            acc[nt] = __builtin_amdgcn_wmma_f32_16x16x32_bf16(
                false, a.v, false, bm.v, (short)0, acc[nt], false, false);
        }
    }
    __syncthreads();   // everyone done reading X before H overwrites smem

    // ---- relu + restage H (bf16) into same LDS ----
#pragma unroll
    for (int nt = 0; nt < 4; ++nt)
#pragma unroll
        for (int r = 0; r < 8; ++r) {
            float h = acc[nt][r]; h = h > 0.0f ? h : 0.0f;
            int m = rq * 16 + lh * 8 + r;
            int n = ch * 64 + nt * 16 + ln;
            smem[m * HS + n] = (bf16_t)h;
        }
    __syncthreads();

    // ---- layer 2 (K = 128) ----
    v8f acc2[4];
#pragma unroll
    for (int nt = 0; nt < 4; ++nt) {
        float bv = b2[ch * 64 + nt * 16 + ln];
#pragma unroll
        for (int j = 0; j < 8; ++j) acc2[nt][j] = bv;
    }
#pragma unroll
    for (int k0 = 0; k0 < 128; k0 += 32) {
        Frag a;
        a.q[0] = *(const uint4*)&smem[am * HS + k0 + lh * 8];
        a.q[1] = *(const uint4*)&smem[am * HS + k0 + 16 + lh * 8];
#pragma unroll
        for (int nt = 0; nt < 4; ++nt) {
            Frag bm;
            const bf16_t* wp = W2t + (size_t)(ch * 64 + nt * 16 + ln) * 128 + k0 + lh * 8;
            bm.q[0] = *(const uint4*)(wp);
            bm.q[1] = *(const uint4*)(wp + 16);
            acc2[nt] = __builtin_amdgcn_wmma_f32_16x16x32_bf16(
                false, a.v, false, bm.v, (short)0, acc2[nt], false, false);
        }
    }

    // ---- residual epilogue (in place; block owns its 64 rows) ----
#pragma unroll
    for (int nt = 0; nt < 4; ++nt)
#pragma unroll
        for (int r = 0; r < 8; ++r) {
            int m = rowBase + rq * 16 + lh * 8 + r;
            int n = ch * 64 + nt * 16 + ln;
            float* p = Y + (size_t)m * D_DIM + n;
            *p = *p + acc2[nt][r];
        }
}

// ------------------------------ V scores -----------------------------------
// out[v] = relu(L.reshape(nv,256) @ V_W1 + b1) @ V_W2 + b2
__global__ __launch_bounds__(256) void v_score_kernel(
        const float* __restrict__ L, const bf16_t* __restrict__ W1t,
        const float* __restrict__ b1, const float* __restrict__ w2,
        const float* __restrict__ b2p, float* __restrict__ out) {
    constexpr int KX = 256, XS = 264, HS = 136;
    __shared__ __align__(16) bf16_t smem[64 * XS];
    const int tid = threadIdx.x;
    const int rowBase = blockIdx.x * 64;

    for (int i = tid; i < 64 * KX; i += 256) {
        int r = i / KX, c = i % KX;
        smem[r * XS + c] = (bf16_t)L[(size_t)(rowBase + r) * KX + c];
    }
    __syncthreads();

    const int lane = tid & 31, wv = tid >> 5;
    const int rq = wv >> 1, ch = wv & 1, lh = lane >> 4, ln = lane & 15;
    const int am = rq * 16 + ln;

    v8f acc[4];
#pragma unroll
    for (int nt = 0; nt < 4; ++nt) {
        float bv = b1[ch * 64 + nt * 16 + ln];
#pragma unroll
        for (int j = 0; j < 8; ++j) acc[nt][j] = bv;
    }
#pragma unroll 2
    for (int k0 = 0; k0 < KX; k0 += 32) {
        Frag a;
        a.q[0] = *(const uint4*)&smem[am * XS + k0 + lh * 8];
        a.q[1] = *(const uint4*)&smem[am * XS + k0 + 16 + lh * 8];
#pragma unroll
        for (int nt = 0; nt < 4; ++nt) {
            Frag bm;
            const bf16_t* wp = W1t + (size_t)(ch * 64 + nt * 16 + ln) * KX + k0 + lh * 8;
            bm.q[0] = *(const uint4*)(wp);
            bm.q[1] = *(const uint4*)(wp + 16);
            acc[nt] = __builtin_amdgcn_wmma_f32_16x16x32_bf16(
                false, a.v, false, bm.v, (short)0, acc[nt], false, false);
        }
    }
    __syncthreads();
#pragma unroll
    for (int nt = 0; nt < 4; ++nt)
#pragma unroll
        for (int r = 0; r < 8; ++r) {
            float h = acc[nt][r]; h = h > 0.0f ? h : 0.0f;
            int m = rq * 16 + lh * 8 + r;
            int n = ch * 64 + nt * 16 + ln;
            smem[m * HS + n] = (bf16_t)h;
        }
    __syncthreads();

    if (tid < 64) {
        float s = b2p[0];
        for (int k = 0; k < 128; ++k)
            s += (float)smem[tid * HS + k] * w2[k];
        out[rowBase + tid] = s;
    }
}

// ------------------------------- launcher ----------------------------------
extern "C" void kernel_launch(void* const* d_in, const int* in_sizes, int n_in,
                              void* d_out, int out_size, void* d_ws, size_t ws_size,
                              hipStream_t stream) {
    const int*   clause_index  = (const int*)d_in[2];
    const int*   literal_index = (const int*)d_in[3];
    const int    n_edges       = in_sizes[2];
    const float* L_init_scale  = (const float*)d_in[4];
    const float* C_init_scale  = (const float*)d_in[5];
    const float* LC_scale      = (const float*)d_in[6];
    const float* CL_scale      = (const float*)d_in[7];
    const float* C_W1 = (const float*)d_in[8];
    const float* C_b1 = (const float*)d_in[9];
    const float* C_W2 = (const float*)d_in[10];
    const float* C_b2 = (const float*)d_in[11];
    const float* L_W1 = (const float*)d_in[12];
    const float* L_b1 = (const float*)d_in[13];
    const float* L_W2 = (const float*)d_in[14];
    const float* L_b2 = (const float*)d_in[15];
    const float* V_W1 = (const float*)d_in[16];
    const float* V_b1 = (const float*)d_in[17];
    const float* V_W2 = (const float*)d_in[18];   // (128,1) -> flat 128
    const float* V_b2 = (const float*)d_in[19];

    // workspace layout
    char* ws = (char*)d_ws;
    const size_t szC  = (size_t)N_CLAUSES * D_DIM * sizeof(float);  // 134 MB
    const size_t szL  = (size_t)N_LITS    * D_DIM * sizeof(float);  //  67 MB
    float* C   = (float*)(ws);
    float* LC  = (float*)(ws + szC);
    float* L   = (float*)(ws + 2 * szC);
    float* CL  = (float*)(ws + 2 * szC + szL);
    char*  wp  = ws + 2 * szC + 2 * szL;
    bf16_t* C_W1t = (bf16_t*)(wp);                  wp += (size_t)128 * 256 * 2;
    bf16_t* C_W2t = (bf16_t*)(wp);                  wp += (size_t)128 * 128 * 2;
    bf16_t* L_W1t = (bf16_t*)(wp);                  wp += (size_t)128 * 384 * 2;
    bf16_t* L_W2t = (bf16_t*)(wp);                  wp += (size_t)128 * 128 * 2;
    bf16_t* V_W1t = (bf16_t*)(wp);

    // init embeddings
    fill_kernel<<<(unsigned)(((size_t)N_LITS * D_DIM + 255) / 256), 256, 0, stream>>>(
        L, L_init_scale, (unsigned long long)N_LITS * D_DIM);
    fill_kernel<<<(unsigned)(((size_t)N_CLAUSES * D_DIM + 255) / 256), 256, 0, stream>>>(
        C, C_init_scale, (unsigned long long)N_CLAUSES * D_DIM);

    // bf16 transposed weights
    wconv_kernel<<<(256 * 128 + 255) / 256, 256, 0, stream>>>(C_W1, C_W1t, 256, 128);
    wconv_kernel<<<(128 * 128 + 255) / 256, 256, 0, stream>>>(C_W2, C_W2t, 128, 128);
    wconv_kernel<<<(384 * 128 + 255) / 256, 256, 0, stream>>>(L_W1, L_W1t, 384, 128);
    wconv_kernel<<<(128 * 128 + 255) / 256, 256, 0, stream>>>(L_W2, L_W2t, 128, 128);
    wconv_kernel<<<(256 * 128 + 255) / 256, 256, 0, stream>>>(V_W1, V_W1t, 256, 128);

    const unsigned edgeBlocks = (unsigned)((n_edges + 7) / 8);

    for (int round = 0; round < N_ROUNDS; ++round) {
        // literal -> clause messages
        hipMemsetAsync(LC, 0, szC, stream);
        edge_scatter_kernel<<<edgeBlocks, 256, 0, stream>>>(
            L, literal_index, clause_index, LC, n_edges);
        // clause update (in place, residual)
        mlp_update_kernel<256, false><<<N_CLAUSES / 64, 256, 0, stream>>>(
            C, LC, LC_scale, C_W1t, C_b1, C_W2t, C_b2);
        // clause -> literal messages
        hipMemsetAsync(CL, 0, szL, stream);
        edge_scatter_kernel<<<edgeBlocks, 256, 0, stream>>>(
            C, clause_index, literal_index, CL, n_edges);
        // literal update (in place, residual, with flip)
        mlp_update_kernel<384, true><<<N_LITS / 64, 256, 0, stream>>>(
            L, CL, CL_scale, L_W1t, L_b1, L_W2t, L_b2);
    }

    // variable scores
    v_score_kernel<<<N_VARS / 64, 256, 0, stream>>>(
        L, V_W1t, V_b1, V_W2, V_b2, (float*)d_out);
}
